// Block_local_10943576670236
// MI455X (gfx1250) — matrix operand order repaired and checked by reading, not compiled, not run on hardware
//
#include <hip/hip_runtime.h>
#include <hip/hip_bf16.h>
#include <math.h>

#define Bn 8
#define Cc 512
#define Nn 2048
#define Hh 2048
#define NB (Bn * Nn)          // 16384 positions
#define KHW 9

#define USE_ASYNC_LDS 1       // gfx1250 global_load_async_to_lds_b128 path

typedef __attribute__((ext_vector_type(16))) __bf16        v16bf;
typedef __attribute__((ext_vector_type(8)))  float         v8f;
typedef __attribute__((ext_vector_type(8)))  unsigned short us8;

union ABf { v16bf v; us8 h[2]; };

__device__ __forceinline__ unsigned short f2bf(float f) {
  union { float f; unsigned u; } v; v.f = f;
  unsigned r = v.u + 0x7FFFu + ((v.u >> 16) & 1u);
  return (unsigned short)(r >> 16);
}

__device__ __forceinline__ float gelu_exact(float x) {
  return 0.5f * x * (1.0f + erff(x * 0.70710678118654752f));
}

#if USE_ASYNC_LDS
// Async global->LDS 16B copy (ASYNCcnt-tracked). LDS aperture: addr[31:0] is
// the DS offset, so truncating the generic shared pointer gives VDST.
__device__ __forceinline__ void async_cp_b128(const unsigned short* gptr,
                                              unsigned short* lptr) {
  unsigned lds = (unsigned)(size_t)lptr;
  asm volatile("global_load_async_to_lds_b128 %0, %1, off"
               :: "v"(lds), "v"((unsigned long long)(size_t)gptr)
               : "memory");
}
__device__ __forceinline__ void async_wait0() {
  asm volatile("s_wait_asynccnt 0x0" ::: "memory");
}
#endif

// ---------------------------------------------------------------------------
// Weight f32 -> bf16 conversion (plain)
// ---------------------------------------------------------------------------
__global__ void cvt_bf16_kernel(const float* __restrict__ in,
                                unsigned short* __restrict__ out, int n) {
  int i = blockIdx.x * 256 + threadIdx.x;
  if (i < n) out[i] = f2bf(in[i]);
}

// ---------------------------------------------------------------------------
// f32 [rows][cols] -> bf16 [cols][rows] (tiled LDS transpose)
// block (32,32), grid (cols/32, rows/32)
// ---------------------------------------------------------------------------
__global__ __launch_bounds__(1024)
void cvt_transpose_bf16_kernel(const float* __restrict__ in,
                               unsigned short* __restrict__ out,
                               int rows, int cols) {
  __shared__ unsigned short t[32][33];
  const int r0 = blockIdx.y * 32, c0 = blockIdx.x * 32;
  const int tx = threadIdx.x, ty = threadIdx.y;
  t[ty][tx] = f2bf(in[(size_t)(r0 + ty) * cols + c0 + tx]);
  __syncthreads();
  out[(size_t)(c0 + ty) * rows + r0 + tx] = t[tx][ty];
}

// ---------------------------------------------------------------------------
// Fused (x + pos_embed) + channel LayerNorm -> bf16 [B,N,C] (pos-major!)
// Also raw key -> bf16 [B,N,C]. block (16 n, 16 c-groups), grid (N/16, B)
// ---------------------------------------------------------------------------
__global__ __launch_bounds__(256)
void ln_prep_kernel(const float* __restrict__ q, const float* __restrict__ k,
                    const float* __restrict__ qe, const float* __restrict__ ke,
                    const float* __restrict__ g, const float* __restrict__ bb,
                    unsigned short* __restrict__ qln,
                    unsigned short* __restrict__ kln,
                    unsigned short* __restrict__ kraw) {
  __shared__ float qt[Cc][17], kt[Cc][17];
  __shared__ unsigned short krt[Cc][17];
  __shared__ float rq0[16][16], rq1[16][16], rk0[16][16], rk1[16][16];
  __shared__ float mqv[16], rqv[16], mkv[16], rkv[16];

  const int tx = threadIdx.x, ty = threadIdx.y;
  const int n = blockIdx.x * 16 + tx;
  const int b = blockIdx.y;
  const size_t base = (size_t)b * Cc * Nn + n;

  float sq = 0.f, sq2 = 0.f, sk = 0.f, sk2 = 0.f;
  for (int c = ty; c < Cc; c += 16) {
    size_t idx = base + (size_t)c * Nn;
    float kvr = k[idx];
    float qv = q[idx] + qe[idx];
    float kv = kvr + ke[idx];
    qt[c][tx] = qv; kt[c][tx] = kv; krt[c][tx] = f2bf(kvr);
    sq += qv; sq2 += qv * qv;
    sk += kv; sk2 += kv * kv;
  }
  rq0[ty][tx] = sq; rq1[ty][tx] = sq2;
  rk0[ty][tx] = sk; rk1[ty][tx] = sk2;
  __syncthreads();
  if (ty == 0) {
    float a = 0.f, b2 = 0.f, c2 = 0.f, d2 = 0.f;
    for (int i = 0; i < 16; ++i) {
      a += rq0[i][tx]; b2 += rq1[i][tx];
      c2 += rk0[i][tx]; d2 += rk1[i][tx];
    }
    float mean_q = a / Cc, var_q = b2 / Cc - mean_q * mean_q;
    float mean_k = c2 / Cc, var_k = d2 / Cc - mean_k * mean_k;
    mqv[tx] = mean_q; rqv[tx] = rsqrtf(var_q + 1e-5f);
    mkv[tx] = mean_k; rkv[tx] = rsqrtf(var_k + 1e-5f);
  }
  __syncthreads();

  const int tid = ty * 16 + tx;
  for (int e = tid; e < 16 * Cc; e += 256) {
    int nl = e >> 9;        // 0..15
    int c = e & (Cc - 1);   // 0..511
    size_t pos = (size_t)b * Nn + blockIdx.x * 16 + nl;
    float gg = g[c], bv = bb[c];
    qln[pos * Cc + c] = f2bf((qt[c][nl] - mqv[nl]) * rqv[nl] * gg + bv);
    kln[pos * Cc + c] = f2bf((kt[c][nl] - mkv[nl]) * rkv[nl] * gg + bv);
    kraw[pos * Cc + c] = krt[c][nl];
  }
}

// ---------------------------------------------------------------------------
// bf16 WMMA GEMM:  D[M,Ntot] = A[M,K] (row-major) * B[Ntot,K] (N-MAJOR)
// Workgroup tile 128(M) x 64(N), 8 waves (4x2), each wave 32x32 via 4 WMMA.
// Double-buffered LDS, async global->LDS staging (pure B128 both operands).
// MODE 0: f32 out = acc + bias[m]        (projections)
// MODE 1: bf16 out = gelu(acc + bias[n]) (MLP up)
// MODE 2: f32 out = acc + bias[n]        (MLP down)
// ---------------------------------------------------------------------------
#define LDA_S 40  // LDS row stride (bf16 elems), 16B-aligned, rotates banks

template <int MODE>
__global__ __launch_bounds__(256)
void gemm_bf16_kernel(const unsigned short* __restrict__ Ag, int lda,
                      const unsigned short* __restrict__ Bg, int ldb,
                      long bstrideB, const float* __restrict__ bias,
                      float* __restrict__ outF,
                      unsigned short* __restrict__ outBF, long bstrideOut,
                      int K, int ldout) {
  __shared__ unsigned short Asm[2][128 * LDA_S];
  __shared__ unsigned short Bsm[2][64 * LDA_S];   // [n][k]

  const int tid = threadIdx.x;
  const int lane = tid & 31;
  const int wid = tid >> 5;
  const int wave_m = wid >> 1;   // 0..3
  const int wave_n = wid & 1;    // 0..1
  const int m0 = blockIdx.y * 128;
  const int n0 = blockIdx.x * 64;

  Bg += (size_t)blockIdx.z * bstrideB;
  const size_t obase = (size_t)blockIdx.z * bstrideOut;

  const int khalf = lane >> 4;    // 0/1
  const int arow = lane & 15;     // M (A) / N (B,D) sub-index

  // staging indices (one-time)
  const int a_row0 = (tid * 2) >> 2;            // chunk row for A chunk 0
  const int a_kc0 = ((tid * 2) & 3) * 8;
  const int a_row1 = (tid * 2 + 1) >> 2;
  const int a_kc1 = ((tid * 2 + 1) & 3) * 8;
  const int b_row = tid >> 2;                   // 0..63 (n)
  const int b_kc = (tid & 3) * 8;               // 0..24 (k)

  v8f acc[2][2];
  for (int i = 0; i < 2; ++i)
    for (int j = 0; j < 2; ++j) acc[i][j] = (v8f)0.0f;

  const int nk = K / 32;

#define STAGE(bufi, k0)                                                        \
  do {                                                                         \
    const unsigned short* ga0 = &Ag[(size_t)(m0 + a_row0) * lda + (k0) + a_kc0];\
    const unsigned short* ga1 = &Ag[(size_t)(m0 + a_row1) * lda + (k0) + a_kc1];\
    const unsigned short* gb = &Bg[(size_t)(n0 + b_row) * ldb + (k0) + b_kc];  \
    unsigned short* la0 = &Asm[bufi][a_row0 * LDA_S + a_kc0];                  \
    unsigned short* la1 = &Asm[bufi][a_row1 * LDA_S + a_kc1];                  \
    unsigned short* lb = &Bsm[bufi][b_row * LDA_S + b_kc];                     \
    ASYNC_OR_SYNC(ga0, la0);                                                   \
    ASYNC_OR_SYNC(ga1, la1);                                                   \
    ASYNC_OR_SYNC(gb, lb);                                                     \
  } while (0)

#if USE_ASYNC_LDS
#define ASYNC_OR_SYNC(g, l) async_cp_b128((g), (l))
#define STAGE_WAIT() async_wait0()
#else
#define ASYNC_OR_SYNC(g, l) (*(us8*)(l) = *(const us8*)(g))
#define STAGE_WAIT()
#endif

  STAGE(0, 0);
  STAGE_WAIT();
  __syncthreads();

  for (int it = 0; it < nk; ++it) {
    const int buf = it & 1;
    if (it + 1 < nk) STAGE(buf ^ 1, (it + 1) * 32);

    // ---- fragments per ISA layout ----
    ABf afr[2], bfr[2];
    for (int i = 0; i < 2; ++i) {
      const unsigned short* ap =
          &Asm[buf][(wave_m * 32 + i * 16 + arow) * LDA_S + khalf * 8];
      afr[i].h[0] = *(const us8*)ap;          // K = khalf*8 + 0..7
      afr[i].h[1] = *(const us8*)(ap + 16);   // K = 16 + khalf*8 + 0..7
    }
    for (int j = 0; j < 2; ++j) {
      const unsigned short* bp =
          &Bsm[buf][(wave_n * 32 + j * 16 + arow) * LDA_S + khalf * 8];
      bfr[j].h[0] = *(const us8*)bp;          // K = khalf*8 + 0..7
      bfr[j].h[1] = *(const us8*)(bp + 16);   // K = 16 + khalf*8 + 0..7
    }
    for (int i = 0; i < 2; ++i)
      for (int j = 0; j < 2; ++j)
        acc[i][j] = __builtin_amdgcn_wmma_f32_16x16x32_bf16(
            false, afr[i].v, false, bfr[j].v, (short)0, acc[i][j], false,
            false);

    STAGE_WAIT();
    __syncthreads();
  }
#undef STAGE
#undef ASYNC_OR_SYNC
#undef STAGE_WAIT

  // ---- epilogue: C/D layout (lane 0-15: N=lane,M=r; 16-31: N=lane-16,M=r+8)
  const int msel_off = (lane >> 4) * 8;
  const int nsel = lane & 15;
  for (int i = 0; i < 2; ++i)
    for (int j = 0; j < 2; ++j) {
      int mbase = m0 + wave_m * 32 + i * 16 + msel_off;
      int nn = n0 + wave_n * 32 + j * 16 + nsel;
      for (int r = 0; r < 8; ++r) {
        int m = mbase + r;
        float val = acc[i][j][r];
        if (MODE == 0) {
          val += bias[m];
          outF[obase + (size_t)m * ldout + nn] = val;
        } else if (MODE == 1) {
          val = gelu_exact(val + bias[nn]);
          outBF[obase + (size_t)m * ldout + nn] = f2bf(val);
        } else {
          val += bias[nn];
          outF[obase + (size_t)m * ldout + nn] = val;
        }
      }
    }
}

// ---------------------------------------------------------------------------
// Windowed correlation + softmax*scale -> attnw [B*N, 9]
// ---------------------------------------------------------------------------
__global__ __launch_bounds__(256)
void corr_softmax_kernel(const float* __restrict__ qp,
                         const float* __restrict__ kp,
                         float* __restrict__ attnw) {
  int pos = blockIdx.x * 256 + threadIdx.x;
  int b = pos / Nn;
  int n = pos - b * Nn;
  const float* qb = qp + (size_t)b * Cc * Nn;
  const float* kb = kp + (size_t)b * Cc * Nn;
  float w[KHW];
  for (int d = 0; d < KHW; ++d) w[d] = 0.f;
  for (int c = 0; c < Cc; ++c) {
    float qv = qb[(size_t)c * Nn + n];
    const float* kr = kb + (size_t)c * Nn;
    for (int d = 0; d < KHW; ++d) {
      int nn = n + d - 4;
      if (nn >= 0 && nn < Nn) w[d] += qv * kr[nn];
    }
  }
  float m = w[0];
  for (int d = 1; d < KHW; ++d) m = fmaxf(m, w[d]);
  float s = 0.f;
  for (int d = 0; d < KHW; ++d) { w[d] = __expf(w[d] - m); s += w[d]; }
  const float scale = 0.044194173824159216f;  // 512^-0.5
  float inv = scale / s;
  for (int d = 0; d < KHW; ++d) attnw[(size_t)pos * KHW + d] = w[d] * inv;
}

// ---------------------------------------------------------------------------
// Weighted aggregation + residual + pre-LN2 -> xres f32 [B,C,N], h bf16 [pos,C]
// block (16 n, 16 c-groups), grid (N/16, B)
// ---------------------------------------------------------------------------
__global__ __launch_bounds__(256)
void attn_apply_ln2_kernel(const float* __restrict__ vp,
                           const float* __restrict__ attnw,
                           const float* __restrict__ query,
                           const float* __restrict__ g2,
                           const float* __restrict__ bb2,
                           float* __restrict__ xres,
                           unsigned short* __restrict__ hout) {
  __shared__ float xt[Cc][17];
  __shared__ float rs[16][16], rs2[16][16];
  __shared__ float lm[16], lr[16];

  const int tx = threadIdx.x, ty = threadIdx.y;
  const int n = blockIdx.x * 16 + tx;
  const int b = blockIdx.y;
  const size_t cb = (size_t)b * Cc * Nn;

  float wrow[KHW];
  for (int d = 0; d < KHW; ++d)
    wrow[d] = attnw[((size_t)b * Nn + n) * KHW + d];

  float s = 0.f, s2 = 0.f;
  for (int c = ty; c < Cc; c += 16) {
    float a = 0.f;
    const float* vr = vp + cb + (size_t)c * Nn;
    for (int d = 0; d < KHW; ++d) {
      int nn = n + d - 4;
      if (nn >= 0 && nn < Nn) a += wrow[d] * vr[nn];
    }
    float x = query[cb + (size_t)c * Nn + n] + a;
    xt[c][tx] = x;
    xres[cb + (size_t)c * Nn + n] = x;
    s += x; s2 += x * x;
  }
  rs[ty][tx] = s; rs2[ty][tx] = s2;
  __syncthreads();
  if (ty == 0) {
    float a = 0.f, q = 0.f;
    for (int i = 0; i < 16; ++i) { a += rs[i][tx]; q += rs2[i][tx]; }
    float mean = a / Cc, var = q / Cc - mean * mean;
    lm[tx] = mean; lr[tx] = rsqrtf(var + 1e-5f);
  }
  __syncthreads();

  const int tid = ty * 16 + tx;
  for (int e = tid; e < 16 * Cc; e += 256) {
    int nl = e >> 9;
    int c = e & (Cc - 1);
    float x = xt[c][nl];
    float hv = (x - lm[nl]) * lr[nl] * g2[c] + bb2[c];
    size_t pos = (size_t)b * Nn + blockIdx.x * 16 + nl;
    hout[pos * Cc + c] = f2bf(hv);
  }
}

// ---------------------------------------------------------------------------
// out[b,c,n] = xres[b,c,n] + mlp2[(b*N+n)*C + c]   (LDS-tiled transpose add)
// ---------------------------------------------------------------------------
__global__ __launch_bounds__(1024)
void final_add_kernel(const float* __restrict__ xres,
                      const float* __restrict__ mlp2,
                      float* __restrict__ out) {
  __shared__ float t[32][33];
  const int b = blockIdx.z;
  const int n0 = blockIdx.x * 32;
  const int c0 = blockIdx.y * 32;
  const int tx = threadIdx.x, ty = threadIdx.y;
  t[ty][tx] = mlp2[((size_t)b * Nn + n0 + ty) * Cc + c0 + tx];
  __syncthreads();
  size_t o = (size_t)b * Cc * Nn + (size_t)(c0 + ty) * Nn + n0 + tx;
  out[o] = xres[o] + t[tx][ty];
}

// ---------------------------------------------------------------------------
extern "C" void kernel_launch(void* const* d_in, const int* in_sizes, int n_in,
                              void* d_out, int out_size, void* d_ws,
                              size_t ws_size, hipStream_t stream) {
  const float* query = (const float*)d_in[0];
  const float* key = (const float*)d_in[1];
  const float* query_embed = (const float*)d_in[2];
  const float* key_embed = (const float*)d_in[3];
  const float* wq = (const float*)d_in[4];
  const float* bq = (const float*)d_in[5];
  const float* wk = (const float*)d_in[6];
  const float* bk = (const float*)d_in[7];
  const float* wv = (const float*)d_in[8];
  const float* bv = (const float*)d_in[9];
  const float* g_norm = (const float*)d_in[10];
  const float* b_norm = (const float*)d_in[11];
  const float* g_norm2 = (const float*)d_in[12];
  const float* b_norm2 = (const float*)d_in[13];
  const float* w1 = (const float*)d_in[14];
  const float* b1 = (const float*)d_in[15];
  const float* w2 = (const float*)d_in[16];
  const float* b2 = (const float*)d_in[17];
  float* out = (float*)d_out;

  const size_t SB = (size_t)Bn * Cc * Nn;  // 8,388,608

  char* ws = (char*)d_ws;
  size_t o_qln = 0;                         // bf16 [B,N,C]
  size_t o_kln = o_qln + SB * 2;            // bf16 [B,N,C]
  size_t o_keybf = o_kln + SB * 2;          // bf16 [B,N,C]
  size_t o_wbf = o_keybf + SB * 2;
  size_t o_wq = o_wbf;
  size_t o_wk = o_wq + (size_t)Cc * Cc * 2;
  size_t o_wv = o_wk + (size_t)Cc * Cc * 2;
  size_t o_w1 = o_wv + (size_t)Cc * Cc * 2;  // w1T bf16 [H,C]
  size_t o_w2 = o_w1 + (size_t)Cc * Hh * 2;  // w2T bf16 [C,H]
  size_t o_qp = o_w2 + (size_t)Hh * Cc * 2;  // f32 [B,C,N]
  size_t o_kp = o_qp + SB * 4;
  size_t o_vp = o_kp + SB * 4;
  size_t o_aw = o_vp + SB * 4;               // f32 [B*N,9]
  size_t o_xres = o_aw + (size_t)NB * KHW * 4;
  size_t o_h = o_xres + SB * 4;              // bf16 [NB,C]
  size_t o_act = o_qp;                       // bf16 [NB,H]  (reuses qp+kp)
  size_t o_mlp2 = o_vp;                      // f32 [NB,C]   (reuses vp)

  unsigned short* qln = (unsigned short*)(ws + o_qln);
  unsigned short* kln = (unsigned short*)(ws + o_kln);
  unsigned short* keybf = (unsigned short*)(ws + o_keybf);
  unsigned short* wq_bf = (unsigned short*)(ws + o_wq);
  unsigned short* wk_bf = (unsigned short*)(ws + o_wk);
  unsigned short* wv_bf = (unsigned short*)(ws + o_wv);
  unsigned short* w1t = (unsigned short*)(ws + o_w1);
  unsigned short* w2t = (unsigned short*)(ws + o_w2);
  float* qp = (float*)(ws + o_qp);
  float* kp = (float*)(ws + o_kp);
  float* vp = (float*)(ws + o_vp);
  float* aw = (float*)(ws + o_aw);
  float* xres = (float*)(ws + o_xres);
  unsigned short* hbuf = (unsigned short*)(ws + o_h);
  unsigned short* act = (unsigned short*)(ws + o_act);
  float* mlp2 = (float*)(ws + o_mlp2);

  // 1) weights -> bf16 (projection weights row-major; MLP weights transposed)
  cvt_bf16_kernel<<<(Cc * Cc + 255) / 256, 256, 0, stream>>>(wq, wq_bf, Cc * Cc);
  cvt_bf16_kernel<<<(Cc * Cc + 255) / 256, 256, 0, stream>>>(wk, wk_bf, Cc * Cc);
  cvt_bf16_kernel<<<(Cc * Cc + 255) / 256, 256, 0, stream>>>(wv, wv_bf, Cc * Cc);
  cvt_transpose_bf16_kernel<<<dim3(Hh / 32, Cc / 32), dim3(32, 32), 0, stream>>>(
      w1, w1t, Cc, Hh);   // [C,H] -> [H,C]
  cvt_transpose_bf16_kernel<<<dim3(Cc / 32, Hh / 32), dim3(32, 32), 0, stream>>>(
      w2, w2t, Hh, Cc);   // [H,C] -> [C,H]

  // 2) pos-embed + LN -> bf16 activations [B,N,C]
  ln_prep_kernel<<<dim3(Nn / 16, Bn), dim3(16, 16), 0, stream>>>(
      query, key, query_embed, key_embed, g_norm, b_norm, qln, kln, keybf);

  // 3) projections: W[C,C] x X^T per batch (WMMA, B is N-major)
  dim3 gproj(Nn / 64, Cc / 128, Bn);
  gemm_bf16_kernel<0><<<gproj, 256, 0, stream>>>(
      wq_bf, Cc, qln, Cc, (long)Nn * Cc, bq, qp, nullptr, (long)Cc * Nn, Cc, Nn);
  gemm_bf16_kernel<0><<<gproj, 256, 0, stream>>>(
      wk_bf, Cc, kln, Cc, (long)Nn * Cc, bk, kp, nullptr, (long)Cc * Nn, Cc, Nn);
  gemm_bf16_kernel<0><<<gproj, 256, 0, stream>>>(
      wv_bf, Cc, keybf, Cc, (long)Nn * Cc, bv, vp, nullptr, (long)Cc * Nn, Cc, Nn);

  // 4) windowed correlation + softmax*scale
  corr_softmax_kernel<<<NB / 256, 256, 0, stream>>>(qp, kp, aw);

  // 5) aggregation + residual + LN2 -> xres, h(bf16 [pos,C])
  attn_apply_ln2_kernel<<<dim3(Nn / 16, Bn), dim3(16, 16), 0, stream>>>(
      vp, aw, query, g_norm2, b_norm2, xres, hbuf);

  // 6) MLP up: h[NB,C] x w1T[H,C] -> gelu -> act bf16 (WMMA)
  gemm_bf16_kernel<1><<<dim3(Hh / 64, NB / 128, 1), 256, 0, stream>>>(
      hbuf, Cc, w1t, Cc, 0L, b1, nullptr, act, 0L, Cc, Hh);

  // 7) MLP down: act[NB,H] x w2T[C,H] + b2 -> mlp2 f32 (WMMA)
  gemm_bf16_kernel<2><<<dim3(Cc / 64, NB / 128, 1), 256, 0, stream>>>(
      act, Hh, w2t, Hh, 0L, b2, mlp2, nullptr, 0L, Hh, Cc);

  // 8) out = xres + mlp2^T
  final_add_kernel<<<dim3(Nn / 32, Cc / 32, Bn), dim3(32, 32), 0, stream>>>(
      xres, mlp2, out);

  (void)in_sizes; (void)n_in; (void)out_size; (void)ws_size;
}
